// RPContextPrior_23252952941217
// MI455X (gfx1250) — matrix operand order repaired and checked by reading, not compiled, not run on hardware
//
#include <hip/hip_runtime.h>

// ---------------------------------------------------------------------------
// MonoScene RPContextPrior forward for MI455X (gfx1250), bf16 WMMA GEMM-convs
// LDS-staged B operands, compile-time axis/K geometry, fused epilogues.
// Grid is 3-D (Ntile-group, Mtile, batch) so the conv kernel has zero integer
// division -- pure shift/add wave->tile mapping.
// ---------------------------------------------------------------------------

typedef __attribute__((ext_vector_type(16))) __bf16 bf16x16;
typedef __attribute__((ext_vector_type(8)))  __bf16 bf16x8;
typedef __attribute__((ext_vector_type(8)))  float  f32x8;

#define BS    4
#define SVOX  16200      // 30*18*30 spatial voxels per batch
#define TN    1013       // ceil(SVOX/16) N-tiles
#define NCOL  129600L    // 8 * SVOX (softmax columns)
#define BN_SCALE 0.9999950000374997f   // 1/sqrt(1+1e-5)

// ---------------- small conversion / elementwise kernels -------------------

__global__ void k_cast_f32_bf16(const float* __restrict__ src,
                                __bf16* __restrict__ dst, int n) {
  int i = blockIdx.x * blockDim.x + threadIdx.x;
  if (i < n) dst[i] = (__bf16)src[i];
}

// [64][64][3] fp32 -> [3][64][64] bf16 (per-tap A matrices)
__global__ void k_cvt3tap(const float* __restrict__ src, __bf16* __restrict__ dst) {
  int i = blockIdx.x * blockDim.x + threadIdx.x;
  if (i >= 3 * 64 * 64) return;
  int t = i / 4096, oi = i % 4096;
  dst[i] = (__bf16)src[oi * 3 + t];
}

__global__ void k_bn_scale(const float* __restrict__ g, float* __restrict__ dst, int n) {
  int i = blockIdx.x * blockDim.x + threadIdx.x;
  if (i < n) dst[i] = g[i] * BN_SCALE;
}

// fp32 input -> bf16 channels [0,256) of the 608-channel concat buffer
__global__ void k_input_to_cat(const float* __restrict__ in, __bf16* __restrict__ cat) {
  long i = (long)blockIdx.x * blockDim.x + threadIdx.x;
  if (i >= (long)BS * 256 * SVOX) return;
  long b = i / (256L * SVOX), rem = i % (256L * SVOX);
  cat[b * 608L * SVOX + rem] = (__bf16)in[i];
}

// zero the x_context channels [256,352) of cat (scatter target)
__global__ void k_zero_ctx(__bf16* __restrict__ cat) {
  long i = (long)blockIdx.x * blockDim.x + threadIdx.x;
  if (i >= (long)BS * 96 * SVOX) return;
  long b = i / (96L * SVOX), rem = i % (96L * SVOX);
  cat[b * 608L * SVOX + 256L * SVOX + rem] = (__bf16)0.0f;
}

// softmax over 12 relation classes, T = 0.03; logits fp32 [BS][96][SVOX]
__global__ void k_softmax_rel(const float* __restrict__ logits,
                              __bf16* __restrict__ P /* [BS][12][NCOL] */) {
  long i = (long)blockIdx.x * blockDim.x + threadIdx.x;
  if (i >= (long)BS * NCOL) return;
  int  b = (int)(i / NCOL);
  long j = i % NCOL;                       // j = pos*SVOX + s
  const float* base = logits + (long)b * 96 * SVOX + j;   // stride NCOL per rel
  float x[12], m = -3.4e38f;
  #pragma unroll
  for (int r = 0; r < 12; ++r) { x[r] = base[(long)r * NCOL]; m = fmaxf(m, x[r]); }
  float s = 0.0f;
  #pragma unroll
  for (int r = 0; r < 12; ++r) { x[r] = __expf((x[r] - m) * (1.0f / 0.03f)); s += x[r]; }
  float inv = 1.0f / s;
  __bf16* pb = P + (long)b * 12 * NCOL + j;
  #pragma unroll
  for (int r = 0; r < 12; ++r) pb[(long)r * NCOL] = (__bf16)(x[r] * inv);
}

// x_context[:, lin_c] = P[:, lin_p]  (scatter into cat channels [256,352))
__global__ void k_scatter_ctx(const __bf16* __restrict__ P,
                              const int* __restrict__ map_c,
                              const int* __restrict__ map_p,
                              __bf16* __restrict__ cat) {
  long i = (long)blockIdx.x * blockDim.x + threadIdx.x;
  if (i >= (long)BS * 32768) return;
  int b = (int)(i >> 15), k = (int)(i & 32767);
  const int* mp = map_p + ((long)b * 32768 + k) * 4;   // (h, w, d, pos)
  const int* mc = map_c + ((long)b * 32768 + k) * 4;
  long lin_p = (((long)mp[3] * 30 + mp[0]) * 18 + mp[1]) * 30 + mp[2];
  long lin_c = (((long)mc[3] * 30 + mc[0]) * 18 + mc[1]) * 30 + mc[2];
  int  cpos = (int)(lin_c / SVOX);
  long scol = lin_c % SVOX;
  const __bf16* pb = P + (long)b * 12 * NCOL + lin_p;
  __bf16* ob = cat + (long)b * 608 * SVOX + scol;
  #pragma unroll
  for (int r = 0; r < 12; ++r)
    ob[(long)(256 + r * 8 + cpos) * SVOX] = pb[(long)r * NCOL];
}

// ---------------- fused GEMM-conv with bf16 WMMA ---------------------------
// One wave = one 16(M=outch) x 16(N=voxel) tile. Per 32-K chunk the wave
// stages its 32x16 B tile into a private LDS slot with one coalesced row per
// lane (2 x 16B global loads), then reads the WMMA B layout via ds_load_u16.
// Axis geometry and Cin are compile-time -> magic division + full K unroll.
// Grid: x = N-tile group (8 waves), y = M-tile, z = batch (no index division).
template <int NTAPS, int ASTRIDE, int ALEN, int CIN>
__global__ __launch_bounds__(256)
void k_conv_wmma(const __bf16* __restrict__ act, long act_bstride,
                 const __bf16* __restrict__ wt,        // [NTAPS][Cout][CIN] bf16
                 int Cout, int dil,
                 const float* __restrict__ bnscale, const float* __restrict__ bnbias,
                 const __bf16* __restrict__ res1, long res1_bstride,
                 const __bf16* __restrict__ res2, long res2_bstride,
                 int relu_in, int relu_out,
                 __bf16* __restrict__ out_bf, long out_bstride,
                 float* __restrict__ out_f32, long outf_bstride) {
  __shared__ __bf16 smem[8][2][32][16];          // per-wave double-buffered B tile
  const int lane = threadIdx.x & 31;
  const int wv   = threadIdx.x >> 5;
  const int tn   = blockIdx.x * 8 + wv;
  if (tn >= TN) return;                      // uniform per wave: EXEC stays full
  const int tm   = blockIdx.y;
  const int b    = blockIdx.z;

  const int n    = lane & 15;                // N (voxel) within tile
  const int hv   = lane >> 4;                // wave half
  const int col0 = tn << 4;
  const int col  = col0 + n;
  const bool colok = col < SVOX;
  const int m0   = tm << 4;

  f32x8 acc = {};
  const __bf16* actb = act + (long)b * act_bstride;
  constexpr long MAXB = (long)CIN * SVOX - 16;

  #pragma unroll
  for (int t = 0; t < NTAPS; ++t) {
    const int off = t - (NTAPS >> 1);
    const int sh  = off * dil;
    bool ok;
    if constexpr (NTAPS == 1) {
      ok = colok;
    } else {
      const int ai = (col / ASTRIDE) % ALEN;   // compile-time magic division
      ok = colok && ((unsigned)(ai + sh) < (unsigned)ALEN);
    }
    const long scol0 = (long)col0 + (long)sh * ASTRIDE;
    const __bf16* wtap = wt + (long)t * Cout * CIN;
    #pragma unroll
    for (int k0 = 0; k0 < CIN; k0 += 32) {
      const int si = (k0 >> 5) & 1;
      // ---- stage B tile: lane -> one K row, 16 contiguous columns (32B) ----
      long rbase = (long)(k0 + lane) * SVOX + scol0;
      rbase = rbase < 0 ? 0 : (rbase > MAXB ? MAXB : rbase);  // keep in-buffer
      const __bf16* gp = actb + rbase;
      bf16x8 r0, r1;
      __builtin_memcpy(&r0, gp, 16);           // unaligned-safe 16B loads
      __builtin_memcpy(&r1, gp + 8, 16);
      if (k0 + 32 < CIN)
        __builtin_prefetch(actb + rbase + 32L * SVOX, 0, 0);  // global_prefetch_b8
      *(bf16x8*)&smem[wv][si][lane][0] = r0;   // ds_store_b128
      *(bf16x8*)&smem[wv][si][lane][8] = r1;
      // ---- A (weights): lane M = m0+n; K halves 0..7/16..23 vs 8..15/24..31
      const __bf16* wrow = wtap + (long)(m0 + n) * CIN + k0 + hv * 8;
      bf16x8 a0 = *(const bf16x8*)(wrow);
      bf16x8 a1 = *(const bf16x8*)(wrow + 16);
      bf16x16 A = __builtin_shufflevector(a0, a1, 0, 1, 2, 3, 4, 5, 6, 7,
                                                  8, 9, 10, 11, 12, 13, 14, 15);
      // ---- read B in WMMA layout: lane N = n, K = hv*8 + e (+8 for e >= 8) ----
      bf16x16 B;
      #pragma unroll
      for (int e = 0; e < 16; ++e) {
        float v = ok ? (float)smem[wv][si][hv * 8 + e + (e >= 8 ? 8 : 0)][n] : 0.0f;
        if (relu_in) v = fmaxf(v, 0.0f);
        B[e] = (__bf16)v;
      }
      acc = __builtin_amdgcn_wmma_f32_16x16x32_bf16(
          /*neg_a=*/false, A, /*neg_b=*/false, B,
          /*c_mod=*/(short)0, acc, /*reuse_a=*/false, /*reuse_b=*/false);
    }
  }

  // ---- epilogue: BN, residuals, ReLU, stores. D row m = m0 + hv*8 + r.
  #pragma unroll
  for (int r = 0; r < 8; ++r) {
    const int m = m0 + hv * 8 + r;
    float v = acc[r];
    if (bnscale) v *= bnscale[m];
    if (bnbias)  v += bnbias[m];
    const long rowoff = (long)m * SVOX + col;
    if (res1) v += (float)res1[(long)b * res1_bstride + rowoff];
    if (res2) v += (float)res2[(long)b * res2_bstride + rowoff];
    if (relu_out) v = fmaxf(v, 0.0f);
    if (colok) {
      if (out_bf)  out_bf [(long)b * out_bstride  + rowoff] = (__bf16)v;
      if (out_f32) out_f32[(long)b * outf_bstride + rowoff] = v;
    }
  }
}

// ---------------------------------------------------------------------------

extern "C" void kernel_launch(void* const* d_in, const int* in_sizes, int n_in,
                              void* d_out, int out_size, void* d_ws, size_t ws_size,
                              hipStream_t stream) {
  (void)in_sizes; (void)n_in; (void)out_size; (void)ws_size;
  const long S = SVOX;

  // -------- inputs (setup_inputs dict order; agg_params nested in order) ----
  const float* input    = (const float*)d_in[0];
  const int*   map_ctx  = (const int*)d_in[2];
  const int*   map_Ps   = (const int*)d_in[3];
  auto blkin = [&](int b, int j) { return (const float*)d_in[4 + b * 15 + j]; };
  const float* pred_w   = (const float*)d_in[49];
  const float* pred_b   = (const float*)d_in[50];
  const float* resize_w = (const float*)d_in[51];
  const float* resize_b = (const float*)d_in[52];

  float* out_f   = (float*)d_out;                       // [4][256][S]
  float* plogit  = out_f + 4L * 256 * S;                // [4][96][S] == P_logit

  // -------- workspace layout -----------------------------------------------
  char* ws = (char*)d_ws;
  size_t off = 0;
  auto alloc = [&](size_t bytes) {
    size_t o = off; off += (bytes + 255) & ~(size_t)255; return o;
  };
  __bf16* cat = (__bf16*)(ws + alloc(4UL * 608 * S * 2));  // [x0 | xctx | x3]
  __bf16* xA  = (__bf16*)(ws + alloc(4UL * 256 * S * 2));
  __bf16* xB  = (__bf16*)(ws + alloc(4UL * 256 * S * 2));
  __bf16* o1  = (__bf16*)(ws + alloc(4UL * 64 * S * 2));
  __bf16* o2  = (__bf16*)(ws + alloc(4UL * 64 * S * 2));
  __bf16* o3  = (__bf16*)(ws + alloc(4UL * 64 * S * 2));
  __bf16* o4  = (__bf16*)(ws + alloc(4UL * 64 * S * 2));
  __bf16* Pbf = (__bf16*)(ws + alloc(4UL * 12 * NCOL * 2));
  __bf16* wbf = (__bf16*)(ws + alloc(389120UL * 2));     // all bf16 weights
  float*  scl = (float*)(ws + alloc(1536UL * 4));        // BN-folded scales

  // per-block weight slices: w1(16384) w2t/w3t/w4t(12288) w5(16384) = 69632
  auto w1p = [&](int b) { return wbf + (long)b * 69632; };
  auto w2p = [&](int b) { return wbf + (long)b * 69632 + 16384; };
  auto w3p = [&](int b) { return wbf + (long)b * 69632 + 28672; };
  auto w4p = [&](int b) { return wbf + (long)b * 69632 + 40960; };
  auto w5p = [&](int b) { return wbf + (long)b * 69632 + 53248; };
  __bf16* wpred = wbf + 208896;                           // 24576
  __bf16* wres  = wbf + 233472;                           // 155648
  auto g1s = [&](int b) { return scl + b * 512 + 0; };
  auto g2s = [&](int b) { return scl + b * 512 + 64; };
  auto g3s = [&](int b) { return scl + b * 512 + 128; };
  auto g4s = [&](int b) { return scl + b * 512 + 192; };
  auto g5s = [&](int b) { return scl + b * 512 + 256; };

  // -------- weight / scale repack ------------------------------------------
  for (int b = 0; b < 3; ++b) {
    k_cast_f32_bf16<<<(16384 + 255) / 256, 256, 0, stream>>>(blkin(b, 0), w1p(b), 16384);
    k_cvt3tap<<<(12288 + 255) / 256, 256, 0, stream>>>(blkin(b, 3), w2p(b));
    k_cvt3tap<<<(12288 + 255) / 256, 256, 0, stream>>>(blkin(b, 6), w3p(b));
    k_cvt3tap<<<(12288 + 255) / 256, 256, 0, stream>>>(blkin(b, 9), w4p(b));
    k_cast_f32_bf16<<<(16384 + 255) / 256, 256, 0, stream>>>(blkin(b, 12), w5p(b), 16384);
    k_bn_scale<<<1, 64, 0, stream>>>(blkin(b, 1), g1s(b), 64);
    k_bn_scale<<<1, 64, 0, stream>>>(blkin(b, 4), g2s(b), 64);
    k_bn_scale<<<1, 64, 0, stream>>>(blkin(b, 7), g3s(b), 64);
    k_bn_scale<<<1, 64, 0, stream>>>(blkin(b, 10), g4s(b), 64);
    k_bn_scale<<<1, 256, 0, stream>>>(blkin(b, 13), g5s(b), 256);
  }
  k_cast_f32_bf16<<<(24576 + 255) / 256, 256, 0, stream>>>(pred_w, wpred, 24576);
  k_cast_f32_bf16<<<(155648 + 255) / 256, 256, 0, stream>>>(resize_w, wres, 155648);

  // input -> cat channels [0,256)
  {
    long n = 4L * 256 * S;
    k_input_to_cat<<<(int)((n + 255) / 256), 256, 0, stream>>>(input, cat);
  }

  // -------- conv launchers (compile-time geometry dispatch) -----------------
  const int GX = (TN + 7) / 8;   // N-tile groups per (M-tile, batch)
  auto conv1 = [&](int Ci, const __bf16* a, long abs_, const __bf16* w, int Co,
                   const float* g, const float* bb,
                   const __bf16* r1, long r1b, int rin, int rout,
                   __bf16* ob, long obs, float* of, long ofs) {
    dim3 grid(GX, Co / 16, BS);
    if (Ci == 256)
      k_conv_wmma<1, 1, 1, 256><<<grid, 256, 0, stream>>>(
          a, abs_, w, Co, 0, g, bb, r1, r1b, nullptr, 0, rin, rout, ob, obs, of, ofs);
    else if (Ci == 64)
      k_conv_wmma<1, 1, 1, 64><<<grid, 256, 0, stream>>>(
          a, abs_, w, Co, 0, g, bb, r1, r1b, nullptr, 0, rin, rout, ob, obs, of, ofs);
    else  // 608
      k_conv_wmma<1, 1, 1, 608><<<grid, 256, 0, stream>>>(
          a, abs_, w, Co, 0, g, bb, r1, r1b, nullptr, 0, rin, rout, ob, obs, of, ofs);
  };
  auto conv3 = [&](int axis /*0=d,1=w,2=h*/, const __bf16* a, const __bf16* w,
                   int dil, const float* g, const float* bb,
                   const __bf16* r1, const __bf16* r2, int rin, __bf16* ob) {
    dim3 grid(GX, 4, BS);
    if (axis == 0)
      k_conv_wmma<3, 1, 30, 64><<<grid, 256, 0, stream>>>(
          a, 64L * S, w, 64, dil, g, bb, r1, 64L * S, r2, 64L * S,
          rin, 0, ob, 64L * S, nullptr, 0);
    else if (axis == 1)
      k_conv_wmma<3, 30, 18, 64><<<grid, 256, 0, stream>>>(
          a, 64L * S, w, 64, dil, g, bb, r1, 64L * S, r2, 64L * S,
          rin, 0, ob, 64L * S, nullptr, 0);
    else
      k_conv_wmma<3, 540, 30, 64><<<grid, 256, 0, stream>>>(
          a, 64L * S, w, 64, dil, g, bb, r1, 64L * S, r2, 64L * S,
          rin, 0, ob, 64L * S, nullptr, 0);
  };

  // -------- 3 bottleneck blocks (dil = 1,2,3) ------------------------------
  const __bf16* xin = cat;       long inb  = 608L * S;
  for (int bk = 0; bk < 3; ++bk) {
    const int dil = bk + 1;
    __bf16* xout = (bk == 0) ? xA : (bk == 1) ? xB : (cat + 352L * S);
    long    outb = (bk == 2) ? 608L * S : 256L * S;

    // out1 = relu(bn(conv1x1 C->64))
    conv1(256, xin, inb, w1p(bk), 64, g1s(bk), blkin(bk, 2),
          nullptr, 0, 0, 1, o1, 64L * S, nullptr, 0);
    // out2 = bn(conv 1x1x3 along d, dilated)
    conv3(0, o1, w2p(bk), dil, g2s(bk), blkin(bk, 5), nullptr, nullptr, 0, o2);
    // out3 = bn(conv 1x3x1 along w, relu input) + out2
    conv3(1, o2, w3p(bk), dil, g3s(bk), blkin(bk, 8), o2, nullptr, 1, o3);
    // out4 = bn(conv 3x1x1 along h, relu input) + out2 + out3
    conv3(2, o3, w4p(bk), dil, g4s(bk), blkin(bk, 11), o2, o3, 1, o4);
    // out  = relu(bn(conv1x1 64->C) + x)
    conv1(64, o4, 64L * S, w5p(bk), 256, g5s(bk), blkin(bk, 14),
          xin, inb, 1, 1, xout, outb, nullptr, 0);

    xin = xout; inb = outb;
  }

  // -------- pred conv -> P_logit (fp32, second output) ---------------------
  conv1(256, cat + 352L * S, 608L * S, wpred, 96, nullptr, pred_b,
        nullptr, 0, 0, 0, nullptr, 0, plogit, 96L * S);

  // -------- softmax over relations + scatter into x_context ----------------
  {
    long n = 4L * NCOL;
    k_softmax_rel<<<(int)((n + 255) / 256), 256, 0, stream>>>(plogit, Pbf);
    long nz = 4L * 96 * S;
    k_zero_ctx<<<(int)((nz + 255) / 256), 256, 0, stream>>>(cat);
    long ns = 4L * 32768;
    k_scatter_ctx<<<(int)((ns + 255) / 256), 256, 0, stream>>>(Pbf, map_ctx, map_Ps, cat);
  }

  // -------- resize conv over concat [input | x_context | x] (608 -> 256) ---
  conv1(608, cat, 608L * S, wres, 256, nullptr, resize_b,
        nullptr, 0, 0, 0, nullptr, 0, out_f, 256L * S);
}
